// fuckupGCN_15066745274645
// MI455X (gfx1250) — matrix operand order repaired and checked by reading, not compiled, not run on hardware
//
#include <hip/hip_runtime.h>
#include <hip/hip_bf16.h>

typedef __attribute__((ext_vector_type(16))) _Float16 v16h;
typedef __attribute__((ext_vector_type(8)))  _Float16 v8h;
typedef __attribute__((ext_vector_type(8)))  float    v8f;

#define NEG_SLOPE 0.01f
#define BS_NSTRIDE 40   // halves; B staged N-major [64][40] f16 in LDS (16B-aligned rows)

// NEG_SLOPE < 1  =>  leaky(x) == max(x, NEG_SLOPE*x): one v_max_f32, no cmp/cndmask
__device__ __forceinline__ float leaky_f(float x) { return fmaxf(x, NEG_SLOPE * x); }

// ---------------------------------------------------------------------------
// Fragment layouts for V_WMMA_F32_16X16X32_F16 (wave32), per CDNA5 ISA 7.12.2:
//  A (16x32 f16): lane L holds row M=L%16; half h -> k = (h>>3)*16 + (L>=16?8:0) + (h&7)
//  B (32x16 f16): lane L holds col N=L%16; half h -> k = (h>>3)*16 + (L>=16?8:0) + (h&7)
//  C/D (16x16 f32): lane L holds col N=L%16; VGPR r holds row M = r + (L>=16?8:0)
// ---------------------------------------------------------------------------

// B fragment gathered straight from a global f32 matrix (used once for the tiny w1)
__device__ __forceinline__ v16h load_b_frag_global(const float* __restrict__ B, int ldb,
                                                   int k0, int n0, int K, int N) {
  const int lane = threadIdx.x & 31;
  const int kofs = (lane >> 4) << 3;  // 0 or 8
  const int n = n0 + (lane & 15);
  const bool nok = n < N;
  v16h f{};
#pragma unroll
  for (int h = 0; h < 16; ++h) {
    int k = k0 + ((h >> 3) << 4) + kofs + (h & 7);
    float v = (nok && k < K) ? B[(size_t)k * ldb + n] : 0.f;
    f[h] = (_Float16)v;
  }
  return f;
}

// ---------------------------------------------------------------------------
// Generic WMMA GEMM: C[M,N] = op(A)[M,K] @ B[K,N], optional fused leaky-ReLU.
// Block = 256 threads = 8 waves; block tile = 128(M) x 64(N), wave tile 16x64.
// B k-slab is staged cooperatively into LDS as f16, N-major, so:
//   - global B reads are coalesced float4 and shared by all 8 waves (8x dedup)
//   - f32->f16 conversion happens once per block, not once per wave
//   - each lane's B fragment is two runs of 8 contiguous halves (k, k+16) -> two ds_load_b128
// ---------------------------------------------------------------------------
template <bool TRANS_A, bool LEAKY>
__global__ void __launch_bounds__(256)
wmma_gemm_kernel(const float* __restrict__ A, const float* __restrict__ B,
                 float* __restrict__ C, int M, int N, int K,
                 int lda, int ldb, int ldc) {
  __shared__ _Float16 Bs[64 * BS_NSTRIDE];

  const int wave = threadIdx.x >> 5;
  const int lane = threadIdx.x & 31;
  const int row0 = blockIdx.y * 128 + wave * 16;
  const int col0 = blockIdx.x * 64;

  const int kb   = (lane >> 4) << 3;   // A half-wave k offset: 0 or 8
  const int kofs = (lane >> 4) << 4;   // 0 or 16; unused (B reads also use kb)

  int mrow = row0 + (lane & 15);       // clamp for loads; stores stay guarded
  if (mrow > M - 1) mrow = M - 1;
  const float* arow = A + (size_t)mrow * lda;

  // cooperative B loader: each thread owns 8 cols of one k-row of the 32x64 slab
  const int br = threadIdx.x >> 3;         // k row within slab (0..31)
  const int bc = (threadIdx.x & 7) * 8;    // first col within slab (0..56)

  v8f acc[4] = {};

  for (int k0 = 0; k0 < K; k0 += 32) {
    // ---- stage B[k0:k0+32, col0:col0+64] -> LDS f16, N-major, zero-padded ----
    {
      const int gk = k0 + br;
      const int gn = col0 + bc;
      if (gk < K && gn + 7 < N) {
        const float4 p0 = *(const float4*)(B + (size_t)gk * ldb + gn);
        const float4 p1 = *(const float4*)(B + (size_t)gk * ldb + gn + 4);
        const float v[8] = {p0.x, p0.y, p0.z, p0.w, p1.x, p1.y, p1.z, p1.w};
#pragma unroll
        for (int u = 0; u < 8; ++u)
          Bs[(bc + u) * BS_NSTRIDE + br] = (_Float16)v[u];
      } else {
#pragma unroll
        for (int u = 0; u < 8; ++u) {
          int n = gn + u;
          float x = (gk < K && n < N) ? B[(size_t)gk * ldb + n] : 0.f;
          Bs[(bc + u) * BS_NSTRIDE + br] = (_Float16)x;
        }
      }
    }
    __syncthreads();

    if (k0 + 32 < K) {  // speculative prefetch of next A k-chunk -> global_prefetch_b8
      if (TRANS_A)
        __builtin_prefetch(A + (size_t)(k0 + 32 + (lane & 15)) * lda + mrow, 0, 1);
      else
        __builtin_prefetch(arow + k0 + 32, 0, 1);
    }

    // ---- A fragment ----
    v16h a;
    if (TRANS_A) {
#pragma unroll
      for (int h = 0; h < 16; ++h) {
        int k = k0 + ((h >> 3) << 4) + kb + (h & 7);
        float v = (k < K) ? A[(size_t)k * lda + mrow] : 0.f;
        a[h] = (_Float16)v;
      }
    } else if (k0 + 32 <= K) {   // fast path: two runs of 8 contiguous floats
      const float4 u0 = *(const float4*)(arow + k0 + kb);
      const float4 u1 = *(const float4*)(arow + k0 + kb + 4);
      const float4 u2 = *(const float4*)(arow + k0 + 16 + kb);
      const float4 u3 = *(const float4*)(arow + k0 + 16 + kb + 4);
      const float v[16] = {u0.x, u0.y, u0.z, u0.w, u1.x, u1.y, u1.z, u1.w,
                           u2.x, u2.y, u2.z, u2.w, u3.x, u3.y, u3.z, u3.w};
#pragma unroll
      for (int h = 0; h < 16; ++h) a[h] = (_Float16)v[h];
    } else {                     // tail k-chunk: guarded scalar
#pragma unroll
      for (int h = 0; h < 16; ++h) {
        int k = k0 + ((h >> 3) << 4) + kb + (h & 7);
        float v = (k < K) ? arow[k] : 0.f;
        a[h] = (_Float16)v;
      }
    }

    // ---- 4 B fragments from LDS (2x ds_load_b128 each, no converts) + WMMA ----
#pragma unroll
    for (int t = 0; t < 4; ++t) {
      if (col0 + 16 * t < N) {   // wave-uniform: EXEC all-ones across the WMMA
        const _Float16* bp = &Bs[(t * 16 + (lane & 15)) * BS_NSTRIDE + kb];
        const v8h lo = *(const v8h*)(bp);
        const v8h hi = *(const v8h*)(bp + 16);
        v16h b;
#pragma unroll
        for (int h = 0; h < 8; ++h) { b[h] = lo[h]; b[h + 8] = hi[h]; }
        acc[t] = __builtin_amdgcn_wmma_f32_16x16x32_f16(
            false, a, false, b, (short)0, acc[t], false, false);
      }
    }
    __syncthreads();
  }

  // ---- store (coalesced across lanes 0-15 / 16-31 per row) ----
  const int nloc = lane & 15;
  const int mb = row0 + ((lane >> 4) << 3);
#pragma unroll
  for (int t = 0; t < 4; ++t) {
    int n = col0 + 16 * t + nloc;
    if (n < N) {
#pragma unroll
      for (int r = 0; r < 8; ++r) {
        int m = mb + r;
        if (m < M) {
          float v = acc[t][r];
          if (LEAKY) v = leaky_f(v);
          C[(size_t)m * ldc + n] = v;
        }
      }
    }
  }
}

// ---------------------------------------------------------------------------
// Fused all-pairs MLP: for each (i,j)
//   h1 = leaky(h0_pre[i] - h0_cur[j] + b0)   [72]  (layer0 prefactorized)
//   h2 = leaky(h1 @ w1 + b1)                 [36]  <- 9x v_wmma per 16 pairs
//   s  = h2 @ w2 + b2                        [2]
// One block per i; 8 waves x 8 j-tiles of 16 pairs. w1 fragments resident in
// VGPRs; A fragments built in-register from float4 loads of h0 rows.
// ---------------------------------------------------------------------------
__global__ void __launch_bounds__(256)
pair_mlp_kernel(const float* __restrict__ h0_pre, const float* __restrict__ h0_cur,
                const float* __restrict__ b0, const float* __restrict__ w1,
                const float* __restrict__ b1, const float* __restrict__ w2,
                const float* __restrict__ b2, float* __restrict__ score,
                float* __restrict__ adj_out) {
  __shared__ float lds_h2[8][16][48];
  __shared__ float w2s[72];
  __shared__ float b1s[36];
  __shared__ float b2s[2];

  const int i    = blockIdx.x;
  const int wave = threadIdx.x >> 5;
  const int lane = threadIdx.x & 31;
  const int kb   = (lane >> 4) << 3;

  if (threadIdx.x < 72) w2s[threadIdx.x] = w2[threadIdx.x];
  if (threadIdx.x < 36) b1s[threadIdx.x] = b1[threadIdx.x];
  if (threadIdx.x < 2)  b2s[threadIdx.x] = b2[threadIdx.x];

  // w1 [72x36] padded to 96x48: 3 k-chunks x 3 n-tiles of resident B fragments
  v16h Bf[3][3];
#pragma unroll
  for (int c = 0; c < 3; ++c)
#pragma unroll
    for (int t = 0; t < 3; ++t)
      Bf[c][t] = load_b_frag_global(w1, 36, c * 32, t * 16, 72, 36);

  // hpb = h0_pre[i] + b0 gathered into this lane's A-fragment slots
  float hpb[3][16];
#pragma unroll
  for (int c = 0; c < 3; ++c)
#pragma unroll
    for (int h = 0; h < 16; ++h) {
      int k = c * 32 + ((h >> 3) << 4) + kb + (h & 7);
      hpb[c][h] = (k < 72) ? (h0_pre[(size_t)i * 72 + k] + b0[k]) : 0.f;
    }

  __syncthreads();  // w2s/b1s/b2s visible

  for (int it = 0; it < 8; ++it) {           // uniform trip count for all waves
    const int jt = wave + it * 8;
    const int j0 = jt * 16;
    const float* curRow = h0_cur + (size_t)(j0 + (lane & 15)) * 72;

    v16h a[3];
#pragma unroll
    for (int c = 0; c < 2; ++c) {            // full k-chunks: vectorized b128 loads
      const float4 u0 = *(const float4*)(curRow + c * 32 + kb);
      const float4 u1 = *(const float4*)(curRow + c * 32 + kb + 4);
      const float4 u2 = *(const float4*)(curRow + c * 32 + 16 + kb);
      const float4 u3 = *(const float4*)(curRow + c * 32 + 16 + kb + 4);
      const float hc[16] = {u0.x, u0.y, u0.z, u0.w, u1.x, u1.y, u1.z, u1.w,
                            u2.x, u2.y, u2.z, u2.w, u3.x, u3.y, u3.z, u3.w};
#pragma unroll
      for (int h = 0; h < 16; ++h)
        a[c][h] = (_Float16)leaky_f(hpb[c][h] - hc[h]);
    }
    {                                        // tail chunk: only k=64..71 live
#pragma unroll
      for (int h = 0; h < 16; ++h) {
        int k = 64 + ((h >> 3) << 4) + kb + (h & 7);
        float d = (k < 72) ? (hpb[2][h] - curRow[k]) : 0.f;
        a[2][h] = (_Float16)leaky_f(d);
      }
    }

    v8f acc[3] = {};
#pragma unroll
    for (int t = 0; t < 3; ++t)
#pragma unroll
      for (int c = 0; c < 3; ++c)
        acc[t] = __builtin_amdgcn_wmma_f32_16x16x32_f16(
            false, a[c], false, Bf[c][t], (short)0, acc[t], false, false);

    // h2 = leaky(acc + b1) -> LDS (per-wave slice)
    const int nn = lane & 15;
    const int mb = (lane >> 4) << 3;
#pragma unroll
    for (int t = 0; t < 3; ++t) {
      int n = t * 16 + nn;
      float bias = (n < 36) ? b1s[n] : 0.f;
#pragma unroll
      for (int r = 0; r < 8; ++r)
        lds_h2[wave][mb + r][n] = leaky_f(acc[t][r] + bias);
    }
    __syncthreads();

    // tiny 36->2 head: lanes 0..15 each own one pair
    if (lane < 16) {
      float s0 = b2s[0], s1 = b2s[1];
#pragma unroll 4
      for (int n = 0; n < 36; ++n) {
        float h = lds_h2[wave][lane][n];
        s0 += h * w2s[2 * n + 0];
        s1 += h * w2s[2 * n + 1];
      }
      size_t p = (size_t)i * 1024 + (size_t)(j0 + lane);
      score[2 * p + 0] = s0;
      score[2 * p + 1] = s1;
      adj_out[p] = s1;
    }
    __syncthreads();
  }
}

// ---------------------------------------------------------------------------
extern "C" void kernel_launch(void* const* d_in, const int* in_sizes, int n_in,
                              void* d_out, int out_size, void* d_ws, size_t ws_size,
                              hipStream_t stream) {
  (void)in_sizes; (void)n_in; (void)out_size; (void)ws_size;

  const float* pre    = (const float*)d_in[0];
  const float* cur    = (const float*)d_in[1];
  const float* adj    = (const float*)d_in[2];
  const float* gc1_w  = (const float*)d_in[3];
  const float* gc2_w  = (const float*)d_in[4];
  const float* fc1_w0 = (const float*)d_in[5];
  const float* fc1_b0 = (const float*)d_in[6];
  const float* fc1_w1 = (const float*)d_in[7];
  const float* fc1_b1 = (const float*)d_in[8];
  const float* fc1_w2 = (const float*)d_in[9];
  const float* fc1_b2 = (const float*)d_in[10];
  const float* fc2_w0 = (const float*)d_in[11];
  const float* fc2_b0 = (const float*)d_in[12];
  const float* fc2_w1 = (const float*)d_in[13];
  const float* fc2_b1 = (const float*)d_in[14];
  const float* fc2_w2 = (const float*)d_in[15];
  const float* fc2_b2 = (const float*)d_in[16];

  float* out    = (float*)d_out;
  float* score1 = out;                       // [1024*1024*2]
  float* score2 = out + 2 * 1024 * 1024;     // [1024*1024*2]
  float* adj_o  = out + 4 * 1024 * 1024;     // [1024*1024]; stages round-1 adj too

  float* ws   = (float*)d_ws;                // ~2.9 MB total
  float* preW = ws;                          // [1024*144]
  float* curW = preW + 1024 * 144;
  float* preA = curW + 1024 * 144;
  float* curA = preA + 1024 * 144;
  float* h0p  = curA + 1024 * 144;           // [1024*72]
  float* h0c  = h0p + 1024 * 72;

  const dim3 blk(256);
  auto gg = [](int M, int N) { return dim3((unsigned)((N + 63) / 64), (unsigned)((M + 127) / 128)); };

  // ---------------- round 1 ----------------
  wmma_gemm_kernel<false, false><<<gg(1024, 144), blk, 0, stream>>>(
      pre, gc1_w, preW, 1024, 144, 144, 144, 144, 144);
  wmma_gemm_kernel<false, false><<<gg(1024, 144), blk, 0, stream>>>(
      cur, gc1_w, curW, 1024, 144, 144, 144, 144, 144);
  wmma_gemm_kernel<false, true><<<gg(1024, 144), blk, 0, stream>>>(
      adj, curW, preA, 1024, 144, 1024, 1024, 144, 144);        // leaky(adj @ cur_w)
  wmma_gemm_kernel<true, true><<<gg(1024, 144), blk, 0, stream>>>(
      adj, preW, curA, 1024, 144, 1024, 1024, 144, 144);        // leaky(adj^T @ pre_w)
  wmma_gemm_kernel<false, false><<<gg(1024, 72), blk, 0, stream>>>(
      preA, fc1_w0, h0p, 1024, 72, 144, 144, 72, 72);           // pre @ w0
  wmma_gemm_kernel<false, false><<<gg(1024, 72), blk, 0, stream>>>(
      curA, fc1_w0, h0c, 1024, 72, 144, 144, 72, 72);           // cur @ w0
  pair_mlp_kernel<<<1024, blk, 0, stream>>>(
      h0p, h0c, fc1_b0, fc1_w1, fc1_b1, fc1_w2, fc1_b2, score1, adj_o);

  // ---------------- round 2 ----------------
  wmma_gemm_kernel<false, false><<<gg(1024, 144), blk, 0, stream>>>(
      preA, gc2_w, preW, 1024, 144, 144, 144, 144, 144);
  wmma_gemm_kernel<false, false><<<gg(1024, 144), blk, 0, stream>>>(
      curA, gc2_w, curW, 1024, 144, 144, 144, 144, 144);
  wmma_gemm_kernel<false, true><<<gg(1024, 144), blk, 0, stream>>>(
      adj_o, curW, preA, 1024, 144, 1024, 1024, 144, 144);
  wmma_gemm_kernel<true, true><<<gg(1024, 144), blk, 0, stream>>>(
      adj_o, preW, curA, 1024, 144, 1024, 1024, 144, 144);
  wmma_gemm_kernel<false, false><<<gg(1024, 72), blk, 0, stream>>>(
      preA, fc2_w0, h0p, 1024, 72, 144, 144, 72, 72);
  wmma_gemm_kernel<false, false><<<gg(1024, 72), blk, 0, stream>>>(
      curA, fc2_w0, h0c, 1024, 72, 144, 144, 72, 72);
  pair_mlp_kernel<<<1024, blk, 0, stream>>>(
      h0p, h0c, fc2_b0, fc2_w1, fc2_b1, fc2_w2, fc2_b2, score2, adj_o);
}
